// GNN_28518582846169
// MI455X (gfx1250) — matrix operand order repaired and checked by reading, not compiled, not run on hardware
//
#include <hip/hip_runtime.h>

// ---------------------------------------------------------------------------
// MI455X (gfx1250) GNN forward.
//   * All large GEMMs run on v_wmma_f32_16x16x32_bf16 (wave32, 8 waves/block).
//     Each wave owns a 32-row x 128-col tile: two A fragments re-use every B
//     fragment twice (loads/WMMA = 1.25), B fragments loaded in groups of 4
//     into distinct registers so loads clause-up and WMMAs issue back-to-back.
//   * fp32 master copies of residual streams (x, e); bf16 shadows feed WMMA.
//     Weights transposed to [N x Kp] bf16, K padded to mult of 32 so fragments
//     are contiguous 16B/32B loads per CDNA5 ISA §7.12.2 layouts.
//   * bias+ReLU+LayerNorm+residual fused in-register (shfl_xor butterflies).
//   * Scatter-mean via fp32 atomics; memsets via hipMemsetAsync (capturable).
// ---------------------------------------------------------------------------

typedef __attribute__((ext_vector_type(16))) __bf16 v16bf;
typedef __attribute__((ext_vector_type(8)))  __bf16 v8bf;
typedef __attribute__((ext_vector_type(8)))  float  v8f;

#define D_FEAT 128

// ---------------------------------------------------------------------------
// Weight prep: fp32 W[K x N] -> bf16 Wt[N x Kp] (transposed, zero-padded K).
// ---------------------------------------------------------------------------
__global__ void prep_wt(const float* __restrict__ W, __bf16* __restrict__ Wt,
                        int K, int N, int Kp) {
  int id = blockIdx.x * 256 + threadIdx.x;
  if (id >= N * Kp) return;
  int n = id / Kp, k = id - n * Kp;
  float v = (k < K) ? W[(size_t)k * N + n] : 0.0f;
  Wt[(size_t)n * Kp + k] = (__bf16)v;
}

// ---------------------------------------------------------------------------
// Encoder input prep (bf16, K padded to 32).
// Node encoder input: X with cols 1..15 zeroed -> only col 0 survives.
// ---------------------------------------------------------------------------
__global__ void prep_node_in(const float* __restrict__ X, __bf16* __restrict__ A, int N) {
  int id = blockIdx.x * 256 + threadIdx.x;
  if (id >= N * 32) return;
  int n = id >> 5, c = id & 31;
  A[id] = (c == 0) ? (__bf16)X[(size_t)n * 16] : (__bf16)0.0f;
}

__global__ void prep_edge_in(const float* __restrict__ EA, __bf16* __restrict__ A, int E) {
  int id = blockIdx.x * 256 + threadIdx.x;
  if (id >= E * 32) return;
  int e = id >> 5, c = id & 31;
  A[id] = (c < 15) ? (__bf16)EA[(size_t)e * 15 + c] : (__bf16)0.0f;
}

// ---------------------------------------------------------------------------
// Fused GEMM + bias + (ReLU) + LayerNorm + (residual), bf16 WMMA.
//   A  : bf16 [M x Kp]   (Kp multiple of 32)
//   Wt : bf16 [128 x Kp] (transposed weights)
//   out = resid? (resid + LN(act(A@W+b))) : LN(act(A@W+b))
// 256 threads = 8 waves; wave w owns rows [blk*256 + 32w, +32).
// ---------------------------------------------------------------------------
__global__ __launch_bounds__(256)
void gemm_ln(const __bf16* __restrict__ A, int M, int Kp,
             const __bf16* __restrict__ Wt,
             const float* __restrict__ bias,
             const float* __restrict__ gamma,
             const float* __restrict__ beta,
             const float* __restrict__ resid,   // fp32 [M x 128] or null
             float* __restrict__ out_f32,       // fp32 [M x 128] or null
             __bf16* __restrict__ out_bf,       // bf16 [M x 128] or null
             int relu)
{
  const int lane = threadIdx.x & 31;
  const int wave = threadIdx.x >> 5;
  const int hf   = lane >> 4;     // half-wave: selects K sub-range of fragment
  const int lm   = lane & 15;

  const int row_base = blockIdx.x * 256 + wave * 32;

  // A-fragment rows (lane layout: M = lane%16 in both half-waves). Clamp OOB;
  // stores are guarded below, clamped rows just compute garbage we discard.
  int ar0 = row_base + lm;        if (ar0 >= M) ar0 = M - 1;
  int ar1 = row_base + 16 + lm;   if (ar1 >= M) ar1 = M - 1;
  const __bf16* __restrict__ ap0 = A + (size_t)ar0 * Kp;
  const __bf16* __restrict__ ap1 = A + (size_t)ar1 * Kp;
  const __bf16* __restrict__ bp  = Wt + (size_t)lm * Kp;  // + t*16*Kp per tile

  v8f acc[2][8] = {};             // [row-group][column tile of 16]

  for (int ks = 0; ks < Kp; ks += 32) {
    // A 16x32 bf16 fragment: elems 0..7 = K[hf*8 .. +8), elems 8..15 = +16.
    const int ka = ks + hf * 8;
    v8bf a0lo = *(const v8bf*)(ap0 + ka);
    v8bf a0hi = *(const v8bf*)(ap0 + ka + 16);
    v8bf a1lo = *(const v8bf*)(ap1 + ka);
    v8bf a1hi = *(const v8bf*)(ap1 + ka + 16);
    v16bf af0 = __builtin_shufflevector(a0lo, a0hi,
        0, 1, 2, 3, 4, 5, 6, 7, 8, 9, 10, 11, 12, 13, 14, 15);
    v16bf af1 = __builtin_shufflevector(a1lo, a1hi,
        0, 1, 2, 3, 4, 5, 6, 7, 8, 9, 10, 11, 12, 13, 14, 15);

    // B 32x16 bf16 fragment: lane holds col n = lane%16, elems 0..7 = K[hf*8 .. +8),
    // elems 8..15 = +16. Load 4 tiles into distinct regs -> 1 wait, 8 WMMAs back-to-back.
    const int kb = ks + hf * 8;
#pragma unroll
    for (int g = 0; g < 2; ++g) {
      v16bf bf[4];
#pragma unroll
      for (int t4 = 0; t4 < 4; ++t4) {
        const __bf16* bq = bp + (size_t)(g * 4 + t4) * 16 * Kp + kb;
        bf[t4] = __builtin_shufflevector(*(const v8bf*)bq, *(const v8bf*)(bq + 16),
            0, 1, 2, 3, 4, 5, 6, 7, 8, 9, 10, 11, 12, 13, 14, 15);
      }
#pragma unroll
      for (int t4 = 0; t4 < 4; ++t4) {
        const int t = g * 4 + t4;
        acc[0][t] = __builtin_amdgcn_wmma_f32_16x16x32_bf16(
            false, af0, false, bf[t4], (short)0, acc[0][t], false, false);
        acc[1][t] = __builtin_amdgcn_wmma_f32_16x16x32_bf16(
            false, af1, false, bf[t4], (short)0, acc[1][t], false, false);
      }
    }
  }

  // ---- Epilogue: bias + relu + LayerNorm over 128 features, in-register ----
  float bs[8], gm[8], bt[8];
#pragma unroll
  for (int t = 0; t < 8; ++t) {
    int n = t * 16 + lm;
    bs[t] = bias[n];
    gm[t] = gamma[n];
    bt[t] = beta[n];
  }

  // C/D layout: VGPR j holds row j (lanes 0-15) or row 8+j (lanes 16-31);
  // column n = t*16 + lane%16. Row sums reduce across the 16-lane half only.
#pragma unroll
  for (int rg = 0; rg < 2; ++rg) {
    float mean[8], rstd[8];
#pragma unroll
    for (int j = 0; j < 8; ++j) {
      float s = 0.0f, q = 0.0f;
#pragma unroll
      for (int t = 0; t < 8; ++t) {
        float v = acc[rg][t][j] + bs[t];
        if (relu) v = fmaxf(v, 0.0f);
        acc[rg][t][j] = v;
        s += v;
        q += v * v;
      }
#pragma unroll
      for (int m = 1; m < 16; m <<= 1) {   // butterfly within 16-lane half
        s += __shfl_xor(s, m, 32);
        q += __shfl_xor(q, m, 32);
      }
      float mu = s * (1.0f / 128.0f);
      mean[j] = mu;
      float var = q * (1.0f / 128.0f) - mu * mu;
      rstd[j] = rsqrtf(var + 1e-5f);
    }

#pragma unroll
    for (int j = 0; j < 8; ++j) {
      int r = row_base + rg * 16 + hf * 8 + j;
      if (r >= M) continue;
#pragma unroll
      for (int t = 0; t < 8; ++t) {
        int n = t * 16 + lm;
        float v = (acc[rg][t][j] - mean[j]) * rstd[j] * gm[t] + bt[t];
        size_t idx = (size_t)r * D_FEAT + n;
        if (resid)   v += resid[idx];
        if (out_f32) out_f32[idx] = v;
        if (out_bf)  out_bf[idx] = (__bf16)v;
      }
    }
  }
}

// ---------------------------------------------------------------------------
// Concat builders (bf16, zero-padded K).
// ---------------------------------------------------------------------------
__global__ void concat_em1(const __bf16* __restrict__ xb, const __bf16* __restrict__ eb,
                           const int* __restrict__ row, const int* __restrict__ col,
                           const float* __restrict__ u, __bf16* __restrict__ A, int E) {
  int e = blockIdx.x;
  if (e >= E) return;
  const __bf16* xr = xb + (size_t)row[e] * D_FEAT;
  const __bf16* xc = xb + (size_t)col[e] * D_FEAT;
  const __bf16* ee = eb + (size_t)e * D_FEAT;
  __bf16* o = A + (size_t)e * 416;
  for (int i = threadIdx.x; i < 416; i += blockDim.x) {
    __bf16 v;
    if (i < 128)      v = xr[i];
    else if (i < 256) v = xc[i - 128];
    else if (i < 384) v = ee[i - 256];
    else if (i == 384) v = (__bf16)u[0];
    else               v = (__bf16)0.0f;
    o[i] = v;
  }
}

__global__ void concat_nm1(const __bf16* __restrict__ xb, const __bf16* __restrict__ eb,
                           const int* __restrict__ row, __bf16* __restrict__ A, int E) {
  int e = blockIdx.x;
  if (e >= E) return;
  const __bf16* xr = xb + (size_t)row[e] * D_FEAT;
  const __bf16* ee = eb + (size_t)e * D_FEAT;
  __bf16* o = A + (size_t)e * 256;
  for (int i = threadIdx.x; i < 256; i += blockDim.x)
    o[i] = (i < 128) ? xr[i] : ee[i - 128];
}

__global__ void concat_nm2(const __bf16* __restrict__ xb, const float* __restrict__ agg,
                           const float* __restrict__ cnt, const float* __restrict__ u,
                           __bf16* __restrict__ A, int N) {
  int n = blockIdx.x;
  if (n >= N) return;
  float inv = 1.0f / fmaxf(cnt[n], 1.0f);
  const __bf16* xr = xb + (size_t)n * D_FEAT;
  const float*  ar = agg + (size_t)n * D_FEAT;
  __bf16* o = A + (size_t)n * 288;
  for (int i = threadIdx.x; i < 288; i += blockDim.x) {
    __bf16 v;
    if (i < 128)       v = xr[i];
    else if (i < 256)  v = (__bf16)(ar[i - 128] * inv);
    else if (i == 256) v = (__bf16)u[0];
    else               v = (__bf16)0.0f;
    o[i] = v;
  }
}

// ---------------------------------------------------------------------------
// Scatter-mean accumulation (fp32 atomics into node-sized agg + counts).
// ---------------------------------------------------------------------------
__global__ void scatter_acc(const float* __restrict__ m, const int* __restrict__ col,
                            float* __restrict__ agg, float* __restrict__ cnt, int E) {
  int e = blockIdx.x;
  if (e >= E) return;
  int c = col[e];
  const float* mr = m + (size_t)e * D_FEAT;
  float* ar = agg + (size_t)c * D_FEAT;
  for (int i = threadIdx.x; i < D_FEAT; i += blockDim.x)
    atomicAdd(ar + i, mr[i]);
  if (threadIdx.x == 0) atomicAdd(cnt + c, 1.0f);
}

// ---------------------------------------------------------------------------
// Final decoder layer: [N x 128] @ [128 x 6] + b  (too thin for WMMA tiles).
// ---------------------------------------------------------------------------
__global__ void dec_final(const __bf16* __restrict__ y, const float* __restrict__ W,
                          const float* __restrict__ b, float* __restrict__ out, int N) {
  int id = blockIdx.x * 256 + threadIdx.x;
  if (id >= N * 6) return;
  int n = id / 6, f = id - n * 6;
  const __bf16* yr = y + (size_t)n * D_FEAT;
  float s = b[f];
#pragma unroll 4
  for (int k = 0; k < D_FEAT; ++k) s += (float)yr[k] * W[(size_t)k * 6 + f];
  out[id] = s;
}

// ---------------------------------------------------------------------------
// Host orchestration.
// ---------------------------------------------------------------------------
extern "C" void kernel_launch(void* const* d_in, const int* in_sizes, int n_in,
                              void* d_out, int out_size, void* d_ws, size_t ws_size,
                              hipStream_t stream) {
  (void)in_sizes; (void)n_in; (void)out_size; (void)ws_size;

  const int N = 10000, E = 160000, MSTEPS = 5;

  const float* X  = (const float*)d_in[0];
  const float* EA = (const float*)d_in[1];
  const int*   ei = (const int*)d_in[2];      // [2 x E]: row = src, col = dst
  const float* u  = (const float*)d_in[3];
  const int* row = ei;
  const int* col = ei + E;

  // Param leaves: JAX pytree order (sorted dict keys: dec, edge_enc, node_enc,
  // proc; lists/tuples in order). 4..13 dec, 14..25 edge_enc, 26..37 node_enc,
  // 38 + 24*s for proc step s (em1 W,b, em1_ln g,b, em2..., nm1..., nm2..., nm3...).
  auto F = [&](int i) -> const float* { return (const float*)d_in[i]; };
  const int DEC = 4, EENC = 14, NENC = 26, PROC = 38, PSTRIDE = 24;

  // ---- workspace carve-up ----
  char* base = (char*)d_ws;
  size_t off = 0;
  auto alloc = [&](size_t bytes) -> void* {
    off = (off + 255) & ~(size_t)255;
    void* p = base + off;
    off += bytes;
    return p;
  };

  float*  x_f   = (float*)alloc((size_t)N * D_FEAT * 4);
  __bf16* x_b   = (__bf16*)alloc((size_t)N * D_FEAT * 2);
  float*  e_f   = (float*)alloc((size_t)E * D_FEAT * 4);
  __bf16* e_b   = (__bf16*)alloc((size_t)E * D_FEAT * 2);
  __bf16* hA    = (__bf16*)alloc((size_t)E * D_FEAT * 2);
  __bf16* hB    = (__bf16*)alloc((size_t)E * D_FEAT * 2);
  float*  m_f   = (float*)alloc((size_t)E * D_FEAT * 4);
  __bf16* A1    = (__bf16*)alloc((size_t)E * 416 * 2);
  __bf16* A2    = (__bf16*)alloc((size_t)E * 256 * 2);
  __bf16* A3    = (__bf16*)alloc((size_t)N * 288 * 2);
  float*  agg   = (float*)alloc((size_t)N * D_FEAT * 4);
  float*  cnt   = (float*)alloc((size_t)N * 4);
  __bf16* An    = (__bf16*)alloc((size_t)N * 32 * 2);
  __bf16* Ae    = (__bf16*)alloc((size_t)E * 32 * 2);

  // ---- bf16 transposed weights (built every call; deterministic) ----
  auto prepW = [&](const float* W, int K, int Kp) -> const __bf16* {
    __bf16* wt = (__bf16*)alloc((size_t)D_FEAT * Kp * 2);
    int tot = D_FEAT * Kp;
    prep_wt<<<dim3((tot + 255) / 256), dim3(256), 0, stream>>>(W, wt, K, D_FEAT, Kp);
    return wt;
  };

  const __bf16 *wt_ne0 = prepW(F(NENC + 0), 16, 32);
  const __bf16 *wt_ne2 = prepW(F(NENC + 4), 128, 128);
  const __bf16 *wt_ne4 = prepW(F(NENC + 8), 128, 128);
  const __bf16 *wt_ee0 = prepW(F(EENC + 0), 15, 32);
  const __bf16 *wt_ee2 = prepW(F(EENC + 4), 128, 128);
  const __bf16 *wt_ee4 = prepW(F(EENC + 8), 128, 128);
  const __bf16 *wt_d0  = prepW(F(DEC + 0), 128, 128);
  const __bf16 *wt_d2  = prepW(F(DEC + 4), 128, 128);
  const __bf16 *wt_em1[MSTEPS], *wt_em2[MSTEPS], *wt_em3[MSTEPS];
  const __bf16 *wt_nm1[MSTEPS], *wt_nm2[MSTEPS], *wt_nm3[MSTEPS];
  for (int s = 0; s < MSTEPS; ++s) {
    int b = PROC + PSTRIDE * s;
    wt_em1[s] = prepW(F(b + 0), 385, 416);
    wt_em2[s] = prepW(F(b + 4), 128, 128);
    wt_em3[s] = prepW(F(b + 8), 128, 128);
    wt_nm1[s] = prepW(F(b + 12), 256, 256);
    wt_nm2[s] = prepW(F(b + 16), 257, 288);
    wt_nm3[s] = prepW(F(b + 20), 128, 128);
  }

  auto gemm = [&](const __bf16* A, int M, int Kp, const __bf16* Wt,
                  const float* bias, const float* g, const float* be,
                  const float* resid, float* of, __bf16* ob, int relu) {
    gemm_ln<<<dim3((M + 255) / 256), dim3(256), 0, stream>>>(
        A, M, Kp, Wt, bias, g, be, resid, of, ob, relu);
  };

  // ---- encoders ----
  prep_node_in<<<dim3((N * 32 + 255) / 256), dim3(256), 0, stream>>>(X, An, N);
  gemm(An, N, 32,  wt_ne0, F(NENC + 1), F(NENC + 2),  F(NENC + 3),  nullptr, nullptr, hA, 1);
  gemm(hA, N, 128, wt_ne2, F(NENC + 5), F(NENC + 6),  F(NENC + 7),  nullptr, nullptr, hB, 1);
  gemm(hB, N, 128, wt_ne4, F(NENC + 9), F(NENC + 10), F(NENC + 11), nullptr, x_f, x_b, 0);

  prep_edge_in<<<dim3((E * 32 + 255) / 256), dim3(256), 0, stream>>>(EA, Ae, E);
  gemm(Ae, E, 32,  wt_ee0, F(EENC + 1), F(EENC + 2),  F(EENC + 3),  nullptr, nullptr, hA, 1);
  gemm(hA, E, 128, wt_ee2, F(EENC + 5), F(EENC + 6),  F(EENC + 7),  nullptr, nullptr, hB, 1);
  gemm(hB, E, 128, wt_ee4, F(EENC + 9), F(EENC + 10), F(EENC + 11), nullptr, e_f, e_b, 0);

  // ---- processor steps ----
  for (int s = 0; s < MSTEPS; ++s) {
    int b = PROC + PSTRIDE * s;

    // EdgeModel: e = e + LN(lin3(LN(relu(lin2(LN(relu(lin1(cat))))))))
    concat_em1<<<dim3(E), dim3(128), 0, stream>>>(x_b, e_b, row, col, u, A1, E);
    gemm(A1, E, 416, wt_em1[s], F(b + 1), F(b + 2),  F(b + 3),  nullptr, nullptr, hA, 1);
    gemm(hA, E, 128, wt_em2[s], F(b + 5), F(b + 6),  F(b + 7),  nullptr, nullptr, hB, 1);
    gemm(hB, E, 128, wt_em3[s], F(b + 9), F(b + 10), F(b + 11), e_f, e_f, e_b, 0);

    // NodeModel: messages + scatter-mean + node MLP + residual
    concat_nm1<<<dim3(E), dim3(128), 0, stream>>>(x_b, e_b, row, A2, E);
    gemm(A2, E, 256, wt_nm1[s], F(b + 13), F(b + 14), F(b + 15), nullptr, m_f, nullptr, 1);

    hipMemsetAsync(agg, 0, (size_t)N * D_FEAT * 4, stream);
    hipMemsetAsync(cnt, 0, (size_t)N * 4, stream);
    scatter_acc<<<dim3(E), dim3(128), 0, stream>>>(m_f, col, agg, cnt, E);

    concat_nm2<<<dim3(N), dim3(128), 0, stream>>>(x_b, agg, cnt, u, A3, N);
    gemm(A3, N, 288, wt_nm2[s], F(b + 17), F(b + 18), F(b + 19), nullptr, nullptr, hA, 1);
    gemm(hA, N, 128, wt_nm3[s], F(b + 21), F(b + 22), F(b + 23), x_f, x_f, x_b, 0);
  }

  // ---- decoder ----
  gemm(x_b, N, 128, wt_d0, F(DEC + 1), F(DEC + 2), F(DEC + 3), nullptr, nullptr, hA, 1);
  gemm(hA,  N, 128, wt_d2, F(DEC + 5), F(DEC + 6), F(DEC + 7), nullptr, nullptr, hB, 1);
  dec_final<<<dim3((N * 6 + 255) / 256), dim3(256), 0, stream>>>(
      hB, F(DEC + 8), F(DEC + 9), (float*)d_out, N);
}
